// MultiHead2DAttention_v2_58763742544531
// MI455X (gfx1250) — matrix operand order repaired and checked by reading, not compiled, not run on hardware
//
#include <hip/hip_runtime.h>

// Problem constants
constexpr int kB  = 2;
constexpr int kL  = 1024;
constexpr int kD  = 1024;
constexpr int kH  = 16;
constexpr int kDH = 64;
constexpr int kM  = 64;       // max relative distance
constexpr int kR  = 129;      // 2*M+1
constexpr int kRKP = 192;     // padded rel_k rows / S_rel stride (3 n-tiles of 64)
constexpr int kRVP = 160;     // padded r-dim for wsum / rel_vT (5 k-steps of 32)
constexpr int kSW  = 144;     // staged S_rel cols in LDS

typedef __attribute__((ext_vector_type(16))) _Float16 v16h;
typedef __attribute__((ext_vector_type(8)))  _Float16 v8h;
typedef __attribute__((ext_vector_type(8)))  float    v8f;
typedef __attribute__((ext_vector_type(4)))  int      v4i;

// CDNA5 async global->LDS copy path (ASYNCcnt), guarded per-toolchain.
#if defined(__has_builtin)
#if __has_builtin(__builtin_amdgcn_global_load_async_to_lds_b128) && \
    __has_builtin(__builtin_amdgcn_s_wait_asynccnt)
#define USE_ASYNC_LDS 1
#endif
#endif
#ifndef USE_ASYNC_LDS
#define USE_ASYNC_LDS 0
#endif

union V16 { v16h v; v8h h[2]; };

__device__ __forceinline__ v8f wmma_f16(const V16& a, const V16& b, v8f c) {
  return __builtin_amdgcn_wmma_f32_16x16x32_f16(false, a.v, false, b.v,
                                                (short)0, c, false, false);
}

// ---------------------------------------------------------------- prep kernels
__global__ void cvt_f16_kernel(const float* __restrict__ in, _Float16* __restrict__ out, int n) {
  int i = blockIdx.x * 256 + threadIdx.x;
  if (i < n) out[i] = (_Float16)in[i];
}

__global__ void transpose_cvt_kernel(const float* __restrict__ in, _Float16* __restrict__ out) {
  int i = blockIdx.x * 256 + threadIdx.x;   // over 1024*1024
  int r = i >> 10, c = i & 1023;
  out[c * kD + r] = (_Float16)in[r * kD + c];
}

__global__ void build_relk_kernel(const float* __restrict__ relk, _Float16* __restrict__ out) {
  int i = blockIdx.x * 256 + threadIdx.x;   // [kRKP][kDH]
  if (i >= kRKP * kDH) return;
  int r = i / kDH, d = i % kDH;
  out[i] = (r < kR) ? (_Float16)relk[r * kDH + d] : (_Float16)0.f;
}

__global__ void build_relvT_kernel(const float* __restrict__ relv, _Float16* __restrict__ out) {
  int i = blockIdx.x * 256 + threadIdx.x;   // [kDH][kRVP]
  if (i >= kDH * kRVP) return;
  int d = i / kRVP, r = i % kRVP;
  out[i] = (r < kR) ? (_Float16)relv[r * kDH + d] : (_Float16)0.f;
}

// ------------------------------------------------------------ generic WMMA GEMM
// C[M,N] = alpha * A[M,K](f16,row-major) * Bt[N,K]^T (f16, n-major rows of K)
// MODE 0: f32 row-major out (ldOut)   MODE 1: f16 head-split [bh][i][d] out
// MODE 3: f16 head-split transposed [bh][d][i] out (packed 8-half stores)
template<int MODE>
__launch_bounds__(256)
__global__ void gemm_wmma_kernel(const _Float16* __restrict__ A,
                                 const _Float16* __restrict__ Bt,
                                 void* __restrict__ out,
                                 int K, int ldA, int ldBt, int ldOut, float alpha)
{
  const int lane = threadIdx.x & 31;
  const int wave = threadIdx.x >> 5;
  const int m0   = blockIdx.x * 128 + wave * 16;
  const int n0   = blockIdx.y * 64;
  const int row  = m0 + (lane & 15);
  const int koff = (lane < 16) ? 0 : 8;     // A-layout half-wave K offset
  const int bko  = 8 * (lane >> 4);         // B-layout half-wave K offset

  v8f acc[4];
  {
    v8f z = {};
#pragma unroll
    for (int s = 0; s < 4; ++s) acc[s] = z;
  }

  for (int kk = 0; kk < K; kk += 32) {
    V16 a;
    const _Float16* ab = A + (size_t)row * ldA + kk + koff;
    a.h[0] = *(const v8h*)(ab);
    a.h[1] = *(const v8h*)(ab + 16);
#pragma unroll
    for (int s = 0; s < 4; ++s) {
      V16 b;
      const _Float16* bb = Bt + (size_t)(n0 + s * 16 + (lane & 15)) * ldBt + kk + bko;
      b.h[0] = *(const v8h*)(bb);
      b.h[1] = *(const v8h*)(bb + 16);
      acc[s] = wmma_f16(a, b, acc[s]);
    }
  }

#pragma unroll
  for (int s = 0; s < 4; ++s) {
    const int cc = n0 + s * 16 + (lane & 15);
    if (MODE == 0) {
      float* O = (float*)out;
#pragma unroll
      for (int g = 0; g < 8; ++g) {
        int rr = m0 + g + 8 * (lane >> 4);
        O[(size_t)rr * ldOut + cc] = acc[s][g] * alpha;
      }
    } else if (MODE == 1) {
      _Float16* O = (_Float16*)out;
      int h = cc >> 6, d = cc & 63;
#pragma unroll
      for (int g = 0; g < 8; ++g) {
        int rr = m0 + g + 8 * (lane >> 4);
        int b  = rr >> 10, i = rr & 1023;
        O[(size_t)((b * kH + h) * kL + i) * kDH + d] = (_Float16)(acc[s][g] * alpha);
      }
    } else { // MODE 3: transposed V
      _Float16* O = (_Float16*)out;
      int h = cc >> 6, d = cc & 63;
      int rr0 = m0 + 8 * (lane >> 4);
      int b = rr0 >> 10, i = rr0 & 1023;
      v8h pack;
#pragma unroll
      for (int g = 0; g < 8; ++g) pack[g] = (_Float16)(acc[s][g] * alpha);
      *(v8h*)(O + (size_t)((b * kH + h) * kDH + d) * kL + i) = pack;
    }
  }
}

// -------------------------------------------------- flash attention + rel bias
__device__ __forceinline__ void compute_s_tiles(v8f sc[4], const V16& a0, const V16& a1,
                                                const _Float16* kbase, const float* srel_s,
                                                int j0, int i0, int mloc, int rowg,
                                                int lane, int bko)
{
#pragma unroll
  for (int s = 0; s < 4; ++s) {
    v8f z = {};
    V16 b0, b1;
    const _Float16* kb = kbase + (size_t)(j0 + s * 16 + (lane & 15)) * kDH;
    b0.h[0] = *(const v8h*)(kb + bko);      b0.h[1] = *(const v8h*)(kb + bko + 16);
    b1.h[0] = *(const v8h*)(kb + 32 + bko); b1.h[1] = *(const v8h*)(kb + 32 + bko + 16);
    z = wmma_f16(a0, b0, z);
    z = wmma_f16(a1, b1, z);
    const int j = j0 + s * 16 + (lane & 15);
#pragma unroll
    for (int g = 0; g < 8; ++g) {
      int il = mloc + rowg + g;                 // local row in [0,64)
      int rr = j - (i0 + il);
      rr = rr < -kM ? -kM : (rr > kM ? kM : rr);
      z[g] += srel_s[il * kSW + rr + kM];
    }
    sc[s] = z;
  }
}

__launch_bounds__(128)
__global__ void flash_rel_kernel(const _Float16* __restrict__ q_h,
                                 const _Float16* __restrict__ k_h,
                                 const _Float16* __restrict__ v_t,
                                 const float* __restrict__ S_rel,
                                 const _Float16* __restrict__ rel_vT,
                                 _Float16* __restrict__ attn_out)
{
  __shared__ float    wsum[64 * kRVP];     // 40 KB, rows wave-private
  __shared__ _Float16 pbuf[4 * 16 * 64];   //  8 KB, one 16x64 p-tile per wave
  extern __shared__ float srel_s[];        // 64 * kSW floats (36 KB, dynamic)

  const int bh   = blockIdx.y;
  const int i0   = blockIdx.x * 64;
  const int tid  = threadIdx.x;
  const int lane = tid & 31;
  const int wave = tid >> 5;

#if USE_ASYNC_LDS
  {
    // Direct cache->LDS copy, tracked by ASYNCcnt (no VGPR round-trip).
    typedef __attribute__((address_space(1))) v4i g_v4i;
    typedef __attribute__((address_space(3))) v4i l_v4i;
    const float* gbase = S_rel + (size_t)(bh * kL + i0) * kRKP;
    constexpr int C4 = kSW / 4;                       // float4s per row
    for (int idx = tid; idx < 64 * C4; idx += 128) {
      int r = idx / C4, c4 = idx % C4;
      __builtin_amdgcn_global_load_async_to_lds_b128(
          (g_v4i*)(gbase + (size_t)r * kRKP + c4 * 4),
          (l_v4i*)(srel_s + idx * 4), 0, 0);
    }
  }
#else
  for (int idx = tid; idx < 64 * kSW; idx += 128) {
    int r = idx / kSW, c = idx % kSW;
    srel_s[idx] = S_rel[(size_t)(bh * kL + i0 + r) * kRKP + c];
  }
#endif
  for (int idx = tid; idx < 64 * kRVP; idx += 128) wsum[idx] = 0.f;
#if USE_ASYNC_LDS
  __builtin_amdgcn_s_wait_asynccnt(0);   // drain async copies before barrier
#endif
  __syncthreads();

  const int mloc = wave * 16;
  const int rowg = 8 * (lane >> 4);
  const int koff = (lane < 16) ? 0 : 8;
  const int bko  = 8 * (lane >> 4);

  V16 a0, a1;   // 16x64 q tile for this wave (2 K-steps of 32)
  {
    const _Float16* qb = q_h + (size_t)(bh * kL + i0 + mloc + (lane & 15)) * kDH;
    a0.h[0] = *(const v8h*)(qb + koff);      a0.h[1] = *(const v8h*)(qb + koff + 16);
    a1.h[0] = *(const v8h*)(qb + 32 + koff); a1.h[1] = *(const v8h*)(qb + 32 + koff + 16);
  }

  const _Float16* kbase = k_h + (size_t)bh * kL * kDH;

  float mcur[8], lsum[8];
#pragma unroll
  for (int g = 0; g < 8; ++g) { mcur[g] = -3.0e38f; lsum[g] = 0.f; }

  // ---------------- pass 1: row max + sum ----------------
  for (int j0 = 0; j0 < kL; j0 += 64) {
    if (j0 + 64 < kL)
      __builtin_prefetch(kbase + (size_t)(j0 + 64 + (lane & 15)) * kDH, 0, 1);
    v8f sc[4];
    compute_s_tiles(sc, a0, a1, kbase, srel_s, j0, i0, mloc, rowg, lane, bko);
#pragma unroll
    for (int g = 0; g < 8; ++g) {
      float vmax = -3.0e38f;
#pragma unroll
      for (int s = 0; s < 4; ++s) vmax = fmaxf(vmax, sc[s][g]);
      vmax = fmaxf(vmax, __shfl_xor(vmax, 1, 32));
      vmax = fmaxf(vmax, __shfl_xor(vmax, 2, 32));
      vmax = fmaxf(vmax, __shfl_xor(vmax, 4, 32));
      vmax = fmaxf(vmax, __shfl_xor(vmax, 8, 32));
      float mn = fmaxf(mcur[g], vmax);
      float part = 0.f;
#pragma unroll
      for (int s = 0; s < 4; ++s) part += __expf(sc[s][g] - mn);
      part += __shfl_xor(part, 1, 32);
      part += __shfl_xor(part, 2, 32);
      part += __shfl_xor(part, 4, 32);
      part += __shfl_xor(part, 8, 32);
      lsum[g] = lsum[g] * __expf(mcur[g] - mn) + part;
      mcur[g] = mn;
    }
  }

  float invl[8];
#pragma unroll
  for (int g = 0; g < 8; ++g) invl[g] = 1.f / lsum[g];

  v8f oacc[4];
  {
    v8f z = {};
#pragma unroll
    for (int s = 0; s < 4; ++s) oacc[s] = z;
  }

  _Float16* pb = pbuf + wave * 16 * 64;

  // ---------------- pass 2: normalized p, o = p*V, wsum ----------------
  for (int j0 = 0; j0 < kL; j0 += 64) {
    v8f sc[4];
    compute_s_tiles(sc, a0, a1, kbase, srel_s, j0, i0, mloc, rowg, lane, bko);
    __syncthreads();                       // previous pbuf consumers done
#pragma unroll
    for (int s = 0; s < 4; ++s) {
      const int j = j0 + s * 16 + (lane & 15);
#pragma unroll
      for (int g = 0; g < 8; ++g) {
        int il = mloc + rowg + g;
        int rr = j - (i0 + il);
        rr = rr < -kM ? -kM : (rr > kM ? kM : rr);
        float p = __expf(sc[s][g] - mcur[g]) * invl[g];
        atomicAdd(&wsum[il * kRVP + rr + kM], p);          // ds_add_f32
        pb[(rowg + g) * 64 + s * 16 + (lane & 15)] = (_Float16)p;
      }
    }
    __syncthreads();
    V16 pa0, pa1;                          // p tile as WMMA A operand
    {
      const _Float16* pbb = pb + (lane & 15) * 64;
      pa0.h[0] = *(const v8h*)(pbb + koff);      pa0.h[1] = *(const v8h*)(pbb + koff + 16);
      pa1.h[0] = *(const v8h*)(pbb + 32 + koff); pa1.h[1] = *(const v8h*)(pbb + 32 + koff + 16);
    }
#pragma unroll
    for (int s = 0; s < 4; ++s) {
      V16 b0, b1;
      const _Float16* vb = v_t + (size_t)(bh * kDH + s * 16 + (lane & 15)) * kL + j0 + bko;
      b0.h[0] = *(const v8h*)(vb);      b0.h[1] = *(const v8h*)(vb + 16);
      b1.h[0] = *(const v8h*)(vb + 32); b1.h[1] = *(const v8h*)(vb + 48);
      oacc[s] = wmma_f16(pa0, b0, oacc[s]);
      oacc[s] = wmma_f16(pa1, b1, oacc[s]);
    }
  }
  __syncthreads();

  // ---------------- o += wsum[16,160] x rel_vT[160,64] ----------------
#pragma unroll
  for (int kk = 0; kk < kRVP; kk += 32) {
    V16 aw;
    const float* wr = &wsum[(mloc + (lane & 15)) * kRVP + kk + koff];
#pragma unroll
    for (int e = 0; e < 8; ++e) aw.v[e] = (_Float16)wr[e];
#pragma unroll
    for (int e = 0; e < 8; ++e) aw.v[8 + e] = (_Float16)wr[16 + e];
#pragma unroll
    for (int s = 0; s < 4; ++s) {
      V16 b;
      const _Float16* rb = rel_vT + (size_t)(s * 16 + (lane & 15)) * kRVP + kk + bko;
      b.h[0] = *(const v8h*)(rb);
      b.h[1] = *(const v8h*)(rb + 16);
      oacc[s] = wmma_f16(aw, b, oacc[s]);
    }
  }

  // ---------------- store combined-head output (f16 [B,L,H*DH]) ----------------
  const int b_ = bh >> 4, h_ = bh & 15;
#pragma unroll
  for (int s = 0; s < 4; ++s) {
    int d = s * 16 + (lane & 15);
#pragma unroll
    for (int g = 0; g < 8; ++g) {
      int i = i0 + mloc + rowg + g;
      attn_out[(size_t)(b_ * kL + i) * kD + h_ * kDH + d] = (_Float16)oacc[s][g];
    }
  }
}

// ---------------------------------------------------------------- host launch
extern "C" void kernel_launch(void* const* d_in, const int* in_sizes, int n_in,
                              void* d_out, int out_size, void* d_ws, size_t ws_size,
                              hipStream_t stream) {
  const float* x    = (const float*)d_in[0];
  const float* Wq   = (const float*)d_in[1];
  const float* Wk   = (const float*)d_in[2];
  const float* Wv   = (const float*)d_in[3];
  const float* Wo   = (const float*)d_in[4];
  const float* relk = (const float*)d_in[5];
  const float* relv = (const float*)d_in[6];
  float* out = (float*)d_out;

  char* ws = (char*)d_ws;
  size_t off = 0;
  auto alloc = [&](size_t bytes) -> void* {
    void* p = ws + off;
    off = (off + bytes + 255) & ~(size_t)255;
    return p;
  };
  const size_t BLD = (size_t)kB * kL * kD;                 // 2M elements
  _Float16* x_h   = (_Float16*)alloc(BLD * 2);
  _Float16* WqT   = (_Float16*)alloc((size_t)kD * kD * 2);
  _Float16* WkT   = (_Float16*)alloc((size_t)kD * kD * 2);
  _Float16* WvT   = (_Float16*)alloc((size_t)kD * kD * 2);
  _Float16* WoT   = (_Float16*)alloc((size_t)kD * kD * 2);
  _Float16* q_h   = (_Float16*)alloc(BLD * 2);
  _Float16* k_h   = (_Float16*)alloc(BLD * 2);
  _Float16* v_t   = (_Float16*)alloc(BLD * 2);
  _Float16* relkP = (_Float16*)alloc((size_t)kRKP * kDH * 2);
  _Float16* relvT = (_Float16*)alloc((size_t)kDH * kRVP * 2);
  float*    S_rel = (float*)   alloc((size_t)kB * kH * kL * kRKP * 4);
  _Float16* aout  = (_Float16*)alloc(BLD * 2);
  (void)ws_size; (void)in_sizes; (void)n_in; (void)out_size;

  const int nBL = kB * kL;                                 // 2048

  cvt_f16_kernel<<<(int)((BLD + 255) / 256), 256, 0, stream>>>(x, x_h, (int)BLD);
  transpose_cvt_kernel<<<(kD * kD) / 256, 256, 0, stream>>>(Wq, WqT);
  transpose_cvt_kernel<<<(kD * kD) / 256, 256, 0, stream>>>(Wk, WkT);
  transpose_cvt_kernel<<<(kD * kD) / 256, 256, 0, stream>>>(Wv, WvT);
  transpose_cvt_kernel<<<(kD * kD) / 256, 256, 0, stream>>>(Wo, WoT);
  build_relk_kernel<<<(kRKP * kDH + 255) / 256, 256, 0, stream>>>(relk, relkP);
  build_relvT_kernel<<<(kDH * kRVP + 255) / 256, 256, 0, stream>>>(relv, relvT);

  dim3 gp(nBL / 128, kD / 64);                             // 16 x 16
  const float qscale = 0.125f;                             // DH^-0.5
  gemm_wmma_kernel<1><<<gp, 256, 0, stream>>>(x_h, WqT, q_h, kD, kD, kD, 0, qscale);
  gemm_wmma_kernel<1><<<gp, 256, 0, stream>>>(x_h, WkT, k_h, kD, kD, kD, 0, 1.0f);
  gemm_wmma_kernel<3><<<gp, 256, 0, stream>>>(x_h, WvT, v_t, kD, kD, kD, 0, 1.0f);

  dim3 gs((kB * kH * kL) / 128, kRKP / 64);                // 256 x 3
  gemm_wmma_kernel<0><<<gs, 256, 0, stream>>>(q_h, relkP, S_rel, kDH, kDH, kDH, kRKP, 1.0f);

  dim3 gf(kL / 64, kB * kH);                               // 16 x 32
  flash_rel_kernel<<<gf, 128, 64 * kSW * sizeof(float), stream>>>(
      q_h, k_h, v_t, S_rel, relvT, aout);

  dim3 go(nBL / 128, kD / 64);                             // 16 x 16
  gemm_wmma_kernel<0><<<go, 256, 0, stream>>>(aout, WoT, out, kD, kD, kD, kD, 1.0f);
}
